// DoubleAttention_39479339385488
// MI455X (gfx1250) — compile-verified, matrix operand order checked by reading, not run on hardware
//
#include <hip/hip_runtime.h>
#include <hip/hip_bf16.h>

typedef __attribute__((ext_vector_type(16))) __bf16 v16bf;
typedef __attribute__((ext_vector_type(8)))  float  v8f;
typedef unsigned int u32x4 __attribute__((ext_vector_type(4)));
typedef int          i32x8 __attribute__((ext_vector_type(8)));
typedef int          i32x4 __attribute__((ext_vector_type(4)));

#define B_ 256
#define H_ 128
#define S_ 2048

// round-to-nearest-even f32 -> bf16
__device__ __forceinline__ unsigned short f32_to_bf16(float f) {
  union { float f; unsigned u; } x; x.f = f;
  unsigned u = x.u;
  unsigned r = (u + 0x7FFFu + ((u >> 16) & 1u)) >> 16;
  return (unsigned short)r;
}

// CDNA5 16-bit WMMA operand layout (ISA 7.12.2, A 16x32 / B 32x16):
// given kk in [0,32): produce lane-half h and 16-bit element slot e in [0,16).
__device__ __forceinline__ void kmap(int kk, int& h, int& e) {
  int top = kk >> 4;       // K 0-15 vs 16-31
  int r   = kk & 15;
  h = r >> 3;              // lane half
  e = (top << 3) + (r & 7);
}

// ---------------------------------------------------------------------------
// Tensor Data Mover: 2D tile load Global -> LDS (ISA 08_async_tensor 7/8).
// D# group0: count=1 | lds_addr | global_addr | type=2.
// D# group1: data_size=4B, tensor_dim0/1, tile_dim0/1, tensor_dim0_stride.
// Groups 2/3 zero (<=2D tensor). Tracked by TENSORcnt.
// This toolchain exposes the 6-arg builtin (g0, g1, g2, g3, extra, cpol).
__device__ __forceinline__ void tdm_load_2d(const void* gsrc, unsigned lds_off,
                                            unsigned tensor_d0, unsigned tensor_d1,
                                            unsigned long long stride0,
                                            unsigned tile_d0, unsigned tile_d1) {
  unsigned long long ga = (unsigned long long)gsrc;
  u32x4 g0;
  g0[0] = 1u;                                               // count=1, user mode
  g0[1] = lds_off;                                          // LDS byte address
  g0[2] = (unsigned)ga;                                     // global_addr[31:0]
  g0[3] = (unsigned)((ga >> 32) & 0x01FFFFFFull) | (2u << 30); // addr[56:32]|type=2
  i32x8 g1;
  g1[0] = (int)(2u << 16);                                  // data_size code 2 = 4B
  g1[1] = (int)((tensor_d0 & 0xFFFFu) << 16);               // tensor_dim0[15:0]
  g1[2] = (int)((tensor_d0 >> 16) | ((tensor_d1 & 0xFFFFu) << 16));
  g1[3] = (int)((tensor_d1 >> 16) | (tile_d0 << 16));       // tile_dim0
  g1[4] = (int)tile_d1;                                     // tile_dim1 (tile_dim2=0)
  g1[5] = (int)(stride0 & 0xFFFFFFFFull);                   // dim0 stride[31:0]
  g1[6] = (int)((stride0 >> 32) & 0xFFFFu);                 // dim0 stride[47:32]
  g1[7] = 0;
  i32x4 gz4 = {0, 0, 0, 0};
  i32x8 gz8 = {0, 0, 0, 0, 0, 0, 0, 0};
  __builtin_amdgcn_tensor_load_to_lds(g0, g1, gz4, gz4, gz8, 0);
}

// ---------------------------------------------------------------------------
// K0: convert W[:, 0:256] and W2[:, 0:128] to bf16 in fragment order so GEMM
//     kernels can load A fragments with coalesced 128-bit loads.
__global__ void prep_frags(const float* __restrict__ W, const float* __restrict__ W2,
                           unsigned short* __restrict__ WabFrag,
                           unsigned short* __restrict__ W2aFrag) {
  int tid = blockIdx.x * blockDim.x + threadIdx.x;
  int nthr = gridDim.x * blockDim.x;
  for (int idx = tid; idx < 128 * 256; idx += nthr) {
    int m = idx >> 8, k = idx & 255;
    int w = m >> 4, mrow = m & 15;
    int ks = k >> 5; int h, e; kmap(k & 31, h, e);
    WabFrag[(((w * 8 + ks) * 32) + (h * 16 + mrow)) * 16 + e] =
        f32_to_bf16(W[m * 384 + k]);
  }
  for (int idx = tid; idx < 128 * 128; idx += nthr) {
    int m = idx >> 7, k = idx & 127;
    int w = m >> 4, mrow = m & 15;
    int ks = k >> 5; int h, e; kmap(k & 31, h, e);
    W2aFrag[(((w * 4 + ks) * 32) + (h * 16 + mrow)) * 16 + e] =
        f32_to_bf16(W2[m * 256 + k]);
  }
}

// ---------------------------------------------------------------------------
// bias[b,h] = sum_k Wm[h, koff+k] * x[b,k]   (rank-1 folding of broadcast ops)
__global__ void bias_kernel(const float* __restrict__ Wm, const float* __restrict__ x,
                            float* __restrict__ outb, int ldw, int koff, int klen) {
  int b = blockIdx.x, h = threadIdx.x;
  float s = 0.f;
  for (int k = 0; k < klen; ++k) s += Wm[h * ldw + koff + k] * x[b * H_ + k];
  outb[b * H_ + h] = s;
}

// ---------------------------------------------------------------------------
// Stage 1: attns[b, s0..s0+63] = v . tanh(Wab @ [static; dynT] + bias1)
// TDM stages f32 tiles into a reused 16KB LDS buffer; threads convert to bf16
// WMMA fragments; 8 waves x (8 K-steps x 4 N-subtiles) bf16 WMMAs.
__global__ __launch_bounds__(256)
void stage1(const float* __restrict__ stat, const float* __restrict__ dyn,
            const unsigned short* __restrict__ WabFrag,
            const float* __restrict__ bias1, const float* __restrict__ v,
            float* __restrict__ attns) {
  __shared__ float ldsRaw[4096];                    // 16 KB TDM landing buffer
  __shared__ unsigned short ldsB[8 * 4 * 32 * 16];  // 32 KB B-fragments
  __shared__ float wpart[8 * 64];                   // per-wave column partials
  int tid  = threadIdx.x;
  int lane = tid & 31, wav = tid >> 5;
  int b  = blockIdx.y;
  int s0 = blockIdx.x * 64;
  unsigned raw_off = (unsigned)(unsigned long long)(void*)ldsRaw;

  // A fragments: this wave's 16x256 strip of Wab, 8 x v16bf registers.
  const v16bf* Ag = reinterpret_cast<const v16bf*>(WabFrag);
  v16bf a[8];
#pragma unroll
  for (int ks = 0; ks < 8; ++ks) a[ks] = Ag[(wav * 8 + ks) * 32 + lane];

  // Phases 0,1: static K-halves. Tile = 64 rows (k) x 64 cols (s), stride S.
  for (int kh = 0; kh < 2; ++kh) {
    if (kh) __syncthreads();            // raw buffer reuse hazard
    if (wav == 0) {
      tdm_load_2d(stat + ((size_t)b * H_ + kh * 64) * S_ + s0, raw_off,
                  S_, H_, S_, /*tile0*/64, /*tile1*/64);
      __builtin_amdgcn_s_wait_tensorcnt(0);
    }
    __syncthreads();
    for (int i = 0; i < 16; ++i) {
      int idx = tid + i * 256;          // idx = krow*64 + n
      int kl = (idx >> 6) + kh * 64, n = idx & 63;
      float val = ldsRaw[idx];
      int ks = kl >> 5; int h, e; kmap(kl & 31, h, e);
      ldsB[(((ks * 4 + (n >> 4)) * 32) + (h * 16 + (n & 15))) * 16 + e] = f32_to_bf16(val);
    }
  }
  // Phases 2,3: dynamic row-halves. Tile = 32 rows (s) x 128 cols (h), stride H.
  for (int nh = 0; nh < 2; ++nh) {
    __syncthreads();
    if (wav == 0) {
      tdm_load_2d(dyn + ((size_t)b * S_ + s0 + nh * 32) * H_, raw_off,
                  H_, S_, H_, /*tile0*/128, /*tile1*/32);
      __builtin_amdgcn_s_wait_tensorcnt(0);
    }
    __syncthreads();
    for (int i = 0; i < 16; ++i) {
      int idx = tid + i * 256;          // idx = nrow*128 + kk2
      int kk2 = idx & 127, n = (idx >> 7) + nh * 32;
      float val = ldsRaw[idx];
      int kl = 128 + kk2; int ks = kl >> 5; int h, e; kmap(kl & 31, h, e);
      ldsB[(((ks * 4 + (n >> 4)) * 32) + (h * 16 + (n & 15))) * 16 + e] = f32_to_bf16(val);
    }
  }
  __syncthreads();

  v8f c[4];
#pragma unroll
  for (int i = 0; i < 4; ++i)
#pragma unroll
    for (int j = 0; j < 8; ++j) c[i][j] = 0.f;

  const v16bf* Bl = reinterpret_cast<const v16bf*>(ldsB);
#pragma unroll
  for (int ks = 0; ks < 8; ++ks) {
#pragma unroll
    for (int ns = 0; ns < 4; ++ns) {
      v16bf bb = Bl[((ks * 4 + ns) * 32) + lane];
      c[ns] = __builtin_amdgcn_wmma_f32_16x16x32_bf16(
          false, a[ks], false, bb, (short)0, c[ns], false, false);
    }
  }

  // Epilogue: tanh + bias, dot with v over this wave's 16 rows.
  int h2 = lane >> 4, nl = lane & 15;
  float vv[8], bb1[8];
#pragma unroll
  for (int r = 0; r < 8; ++r) {
    int mrow = wav * 16 + h2 * 8 + r;   // C layout: lane-half h2 -> rows 8h2+r
    vv[r]  = v[mrow];
    bb1[r] = bias1[b * H_ + mrow];
  }
#pragma unroll
  for (int ns = 0; ns < 4; ++ns) {
    float part = 0.f;
#pragma unroll
    for (int r = 0; r < 8; ++r) part += vv[r] * tanhf(c[ns][r] + bb1[r]);
    part += __shfl_xor(part, 16, 32);           // combine the two lane halves
    if (h2 == 0) wpart[wav * 64 + ns * 16 + nl] = part;
  }
  __syncthreads();
  if (tid < 64) {                               // deterministic tree over waves
    float s = 0.f;
#pragma unroll
    for (int w = 0; w < 8; ++w) s += wpart[w * 64 + tid];
    attns[(size_t)b * S_ + s0 + tid] = s;
  }
}

// ---------------------------------------------------------------------------
// Per-batch: softmax(attns) and context[h] = sum_s softmax * static[b,h,s]
__global__ __launch_bounds__(256)
void softmax_context(const float* __restrict__ stat, const float* __restrict__ attns,
                     float* __restrict__ context) {
  __shared__ float e_lds[S_];  // 8 KB
  __shared__ float red[8];
  int b = blockIdx.x, tid = threadIdx.x, lane = tid & 31, wav = tid >> 5;
  const float* arow = attns + (size_t)b * S_;

  float mx = -__builtin_inff();
  for (int s = tid; s < S_; s += 256) mx = fmaxf(mx, arow[s]);
  for (int o = 16; o; o >>= 1) mx = fmaxf(mx, __shfl_xor(mx, o, 32));
  if (lane == 0) red[wav] = mx;
  __syncthreads();
  mx = red[0];
  for (int w = 1; w < 8; ++w) mx = fmaxf(mx, red[w]);

  float sum = 0.f;
  for (int s = tid; s < S_; s += 256) {
    float e = __expf(arow[s] - mx);
    e_lds[s] = e;
    sum += e;
  }
  for (int o = 16; o; o >>= 1) sum += __shfl_xor(sum, o, 32);
  __syncthreads();                 // red reuse hazard
  if (lane == 0) red[wav] = sum;
  __syncthreads();
  sum = 0.f;
  for (int w = 0; w < 8; ++w) sum += red[w];
  float inv = 1.f / sum;

  // context: wave w handles rows h = 16w..16w+15, lanes stride S (coalesced)
  for (int hr = 0; hr < 16; ++hr) {
    int h = wav * 16 + hr;
    const float* srow = stat + ((size_t)b * H_ + h) * S_;
    float cs = 0.f;
    for (int s = lane; s < S_; s += 32) cs += e_lds[s] * srow[s];
    for (int o = 16; o; o >>= 1) cs += __shfl_xor(cs, o, 32);
    if (lane == 0) context[b * H_ + h] = cs * inv;
  }
}

// ---------------------------------------------------------------------------
// Stage 2: out[b, s0..s0+63] = v2 . tanh(W2a @ static + bias2)   (raw logits)
__global__ __launch_bounds__(256)
void stage2(const float* __restrict__ stat, const unsigned short* __restrict__ W2aFrag,
            const float* __restrict__ bias2, const float* __restrict__ v2,
            float* __restrict__ out) {
  __shared__ float ldsRaw[4096];                    // 16 KB TDM landing buffer
  __shared__ unsigned short ldsB[4 * 4 * 32 * 16];  // 16 KB
  __shared__ float wpart[8 * 64];
  int tid  = threadIdx.x;
  int lane = tid & 31, wav = tid >> 5;
  int b  = blockIdx.y;
  int s0 = blockIdx.x * 64;
  unsigned raw_off = (unsigned)(unsigned long long)(void*)ldsRaw;

  const v16bf* Ag = reinterpret_cast<const v16bf*>(W2aFrag);
  v16bf a[4];
#pragma unroll
  for (int ks = 0; ks < 4; ++ks) a[ks] = Ag[(wav * 4 + ks) * 32 + lane];

  for (int kh = 0; kh < 2; ++kh) {
    if (kh) __syncthreads();
    if (wav == 0) {
      tdm_load_2d(stat + ((size_t)b * H_ + kh * 64) * S_ + s0, raw_off,
                  S_, H_, S_, /*tile0*/64, /*tile1*/64);
      __builtin_amdgcn_s_wait_tensorcnt(0);
    }
    __syncthreads();
    for (int i = 0; i < 16; ++i) {
      int idx = tid + i * 256;
      int kl = (idx >> 6) + kh * 64, n = idx & 63;
      float val = ldsRaw[idx];
      int ks = kl >> 5; int h, e; kmap(kl & 31, h, e);
      ldsB[(((ks * 4 + (n >> 4)) * 32) + (h * 16 + (n & 15))) * 16 + e] = f32_to_bf16(val);
    }
  }
  __syncthreads();

  v8f c[4];
#pragma unroll
  for (int i = 0; i < 4; ++i)
#pragma unroll
    for (int j = 0; j < 8; ++j) c[i][j] = 0.f;

  const v16bf* Bl = reinterpret_cast<const v16bf*>(ldsB);
#pragma unroll
  for (int ks = 0; ks < 4; ++ks) {
#pragma unroll
    for (int ns = 0; ns < 4; ++ns) {
      v16bf bb = Bl[((ks * 4 + ns) * 32) + lane];
      c[ns] = __builtin_amdgcn_wmma_f32_16x16x32_bf16(
          false, a[ks], false, bb, (short)0, c[ns], false, false);
    }
  }

  int h2 = lane >> 4, nl = lane & 15;
  float vv[8], bb2[8];
#pragma unroll
  for (int r = 0; r < 8; ++r) {
    int mrow = wav * 16 + h2 * 8 + r;
    vv[r]  = v2[mrow];
    bb2[r] = bias2[b * H_ + mrow];
  }
#pragma unroll
  for (int ns = 0; ns < 4; ++ns) {
    float part = 0.f;
#pragma unroll
    for (int r = 0; r < 8; ++r) part += vv[r] * tanhf(c[ns][r] + bb2[r]);
    part += __shfl_xor(part, 16, 32);
    if (h2 == 0) wpart[wav * 64 + ns * 16 + nl] = part;
  }
  __syncthreads();
  if (tid < 64) {
    float s = 0.f;
#pragma unroll
    for (int w = 0; w < 8; ++w) s += wpart[w * 64 + tid];
    out[(size_t)b * S_ + s0 + tid] = s;
  }
}

// ---------------------------------------------------------------------------
// Final masked softmax, in place on d_out. mask==true -> -inf (excluded).
__global__ __launch_bounds__(256)
void softmax_mask(float* __restrict__ out, const unsigned char* __restrict__ mask) {
  __shared__ float red[8];
  int b = blockIdx.x, tid = threadIdx.x, lane = tid & 31, wav = tid >> 5;
  float* row = out + (size_t)b * S_;
  const unsigned char* mrow = mask + (size_t)b * S_;

  float vals[8];
  float mx = -__builtin_inff();
#pragma unroll
  for (int i = 0; i < 8; ++i) {
    int s = tid + i * 256;
    float x = mrow[s] ? -__builtin_inff() : row[s];
    vals[i] = x;
    mx = fmaxf(mx, x);
  }
  for (int o = 16; o; o >>= 1) mx = fmaxf(mx, __shfl_xor(mx, o, 32));
  if (lane == 0) red[wav] = mx;
  __syncthreads();
  mx = red[0];
  for (int w = 1; w < 8; ++w) mx = fmaxf(mx, red[w]);

  float sum = 0.f;
#pragma unroll
  for (int i = 0; i < 8; ++i) {
    float e = __expf(vals[i] - mx);
    vals[i] = e;
    sum += e;
  }
  for (int o = 16; o; o >>= 1) sum += __shfl_xor(sum, o, 32);
  __syncthreads();
  if (lane == 0) red[wav] = sum;
  __syncthreads();
  sum = 0.f;
  for (int w = 0; w < 8; ++w) sum += red[w];
  float inv = 1.f / sum;
#pragma unroll
  for (int i = 0; i < 8; ++i) row[tid + i * 256] = vals[i] * inv;
}

// ---------------------------------------------------------------------------
extern "C" void kernel_launch(void* const* d_in, const int* in_sizes, int n_in,
                              void* d_out, int out_size, void* d_ws, size_t ws_size,
                              hipStream_t stream) {
  (void)in_sizes; (void)n_in; (void)out_size; (void)ws_size;
  const float* stat = (const float*)d_in[0];           // [B,H,S]
  const float* dyn  = (const float*)d_in[1];           // [B,S,H]
  const float* dec  = (const float*)d_in[2];           // [B,H,1]
  const unsigned char* mask = (const unsigned char*)d_in[3];  // [B,S] bool
  const float* v  = (const float*)d_in[4];             // [H]
  const float* W  = (const float*)d_in[5];             // [H,3H]
  const float* v2 = (const float*)d_in[6];             // [H]
  const float* W2 = (const float*)d_in[7];             // [H,2H]
  float* out = (float*)d_out;                          // [B,S]

  char* ws = (char*)d_ws;
  unsigned short* WabFrag = (unsigned short*)(ws);              //  64 KB
  unsigned short* W2aFrag = (unsigned short*)(ws + (64 << 10)); //  32 KB
  float* bias1   = (float*)(ws + (96 << 10));                   // 128 KB
  float* bias2   = (float*)(ws + (224 << 10));                  // 128 KB
  float* context = (float*)(ws + (352 << 10));                  // 128 KB
  float* attns   = (float*)(ws + (480 << 10));                  //   2 MB

  prep_frags<<<64, 256, 0, stream>>>(W, W2, WabFrag, W2aFrag);
  bias_kernel<<<B_, 128, 0, stream>>>(W, dec, bias1, 3 * H_, 2 * H_, H_);
  stage1<<<dim3(S_ / 64, B_), 256, 0, stream>>>(stat, dyn, WabFrag, bias1, v, attns);
  softmax_context<<<B_, 256, 0, stream>>>(stat, attns, context);
  bias_kernel<<<B_, 128, 0, stream>>>(W2, context, bias2, 2 * H_, H_, H_);
  stage2<<<dim3(S_ / 64, B_), 256, 0, stream>>>(stat, W2aFrag, bias2, v2, out);
  softmax_mask<<<B_, 256, 0, stream>>>(out, mask);
}
